// LSTM_7078106104551
// MI455X (gfx1250) — compile-verified, matrix operand order checked by reading, not run on hardware
//
#include <hip/hip_runtime.h>
#include <hip/hip_bf16.h>

#define SEQ_    512
#define BATCH_  64
#define NIN_    512
#define NHID_   1024
#define NGATE_  (4 * NHID_)

typedef __bf16 bf16_t;
typedef bf16_t v16bf __attribute__((ext_vector_type(16)));
typedef bf16_t v8bf  __attribute__((ext_vector_type(8)));
typedef float  v8f   __attribute__((ext_vector_type(8)));

// Load a 16-element bf16 WMMA fragment slice: 8 contiguous bf16 at p, 8 at p+16.
// (p is already offset by (lane>=16 ? 8 : 0) per the gfx1250 16-bit A/B layout.)
__device__ __forceinline__ v16bf ldfrag(const bf16_t* p) {
    union { v16bf v; v8bf h[2]; } u;
    u.h[0] = *reinterpret_cast<const v8bf*>(p);
    u.h[1] = *reinterpret_cast<const v8bf*>(p + 16);
    return u.v;
}

__device__ __forceinline__ float sigm_f(float x) {
    return 1.0f / (1.0f + __expf(-x));
}
__device__ __forceinline__ float tanh_f(float x) {
    // 1 - 2/(e^{2x}+1); correct saturation at +/-inf of exp
    return 1.0f - 2.0f / (__expf(2.0f * x) + 1.0f);
}

// ---------------------------------------------------------------------------
// Fused LSTM cell for one layer, one timestep.
//   A = concat(in0 [B x K0], in1 [B x K1]) row-major, bf16
//   Wt = transposed weights [NGATE_ rows x (K0+K1)] row-major bf16 (B operand)
//   act = A @ W + bias ; gates (i,f,o,g) -> c,h update
// Grid: NHID_/16 = 64 blocks x 256 threads (8 waves).
// Wave w: gate g = w>>1, batch-half mhalf = w&1 -> two 16x16 D tiles.
// ---------------------------------------------------------------------------
__global__ __launch_bounds__(256)
void lstm_cell_kernel(const bf16_t* __restrict__ in0, int K0,
                      const bf16_t* __restrict__ in1, int K1,
                      const bf16_t* __restrict__ Wt,
                      const float*  __restrict__ bias,
                      float*        __restrict__ c_state,   // [B][H] in/out
                      bf16_t*       __restrict__ h_bf_out,  // [B][H]
                      float*        __restrict__ h_f32_out) // [B][H]
{
    const int K     = K0 + K1;
    const int n0    = blockIdx.x * 16;        // h-column tile base
    const int tid   = threadIdx.x;
    const int w     = tid >> 5;               // wave 0..7
    const int lane  = tid & 31;
    const int g     = w >> 1;                 // gate 0..3 (i,f,o,g)
    const int mh    = w & 1;                  // batch half: rows [32*mh, 32*mh+32)
    const int half  = lane >> 4;              // K-subgroup select
    const int r     = lane & 15;              // M row (A) / N col (B) within tile

    __shared__ float acts[4][BATCH_][16];     // 16 KB

    v8f acc0 = {};
    v8f acc1 = {};

    // B operand rows of Wt: row = gate*1024 + n0 + r (contiguous K)
    const bf16_t* wrow = Wt + (size_t)(g * NHID_ + n0 + r) * K + half * 8;

    // ---- phase 0: A columns from in0 (k in [0, K0)) ----
    {
        const bf16_t* a0 = in0 + (size_t)(32 * mh + r) * K0 + half * 8;
        const bf16_t* a1 = in0 + (size_t)(32 * mh + 16 + r) * K0 + half * 8;
        for (int k = 0; k < K0; k += 32) {
            v16bf bf = ldfrag(wrow + k);
            v16bf af0 = ldfrag(a0 + k);
            v16bf af1 = ldfrag(a1 + k);
            acc0 = __builtin_amdgcn_wmma_f32_16x16x32_bf16(
                false, af0, false, bf, (short)0, acc0, false, false);
            acc1 = __builtin_amdgcn_wmma_f32_16x16x32_bf16(
                false, af1, false, bf, (short)0, acc1, false, false);
        }
    }
    // ---- phase 1: A columns from in1 (k in [K0, K0+K1)) ----
    {
        const bf16_t* a0 = in1 + (size_t)(32 * mh + r) * K1 + half * 8;
        const bf16_t* a1 = in1 + (size_t)(32 * mh + 16 + r) * K1 + half * 8;
        const bf16_t* wr = wrow + K0;
        for (int k = 0; k < K1; k += 32) {
            v16bf bf = ldfrag(wr + k);
            v16bf af0 = ldfrag(a0 + k);
            v16bf af1 = ldfrag(a1 + k);
            acc0 = __builtin_amdgcn_wmma_f32_16x16x32_bf16(
                false, af0, false, bf, (short)0, acc0, false, false);
            acc1 = __builtin_amdgcn_wmma_f32_16x16x32_bf16(
                false, af1, false, bf, (short)0, acc1, false, false);
        }
    }

    // D tile layout: VGPR v -> M = v + 8*half, N = r
#pragma unroll
    for (int v = 0; v < 8; ++v) {
        acts[g][32 * mh + 8 * half + v][r]      = acc0[v];
        acts[g][32 * mh + 16 + 8 * half + v][r] = acc1[v];
    }
    __syncthreads();

    // Fused elementwise LSTM cell over the [64 x 16] tile
    for (int e = tid; e < BATCH_ * 16; e += 256) {
        const int m   = e >> 4;
        const int n   = e & 15;
        const int col = n0 + n;
        const float xi = acts[0][m][n] + bias[0 * NHID_ + col];
        const float xf = acts[1][m][n] + bias[1 * NHID_ + col];
        const float xo = acts[2][m][n] + bias[2 * NHID_ + col];
        const float xg = acts[3][m][n] + bias[3 * NHID_ + col];

        const float c_old = c_state[m * NHID_ + col];
        const float c_new = sigm_f(xf) * c_old + sigm_f(xi) * tanh_f(xg);
        const float h_new = sigm_f(xo) * tanh_f(c_new);

        c_state[m * NHID_ + col]   = c_new;
        h_bf_out[m * NHID_ + col]  = (bf16_t)h_new;
        h_f32_out[m * NHID_ + col] = h_new;
    }
}

// ---------------------------------------------------------------------------
// One-time (per launch) prep kernels
// ---------------------------------------------------------------------------

// fp32 -> bf16 elementwise (n divisible by 1024); 4 elems/thread
__global__ __launch_bounds__(256)
void cvt_f32_bf16_kernel(const float* __restrict__ src,
                         bf16_t* __restrict__ dst, int n) {
    int i = (blockIdx.x * 256 + threadIdx.x) * 4;
    if (i + 3 < n) {
        float4 v = *reinterpret_cast<const float4*>(src + i);
        dst[i + 0] = (bf16_t)v.x;
        dst[i + 1] = (bf16_t)v.y;
        dst[i + 2] = (bf16_t)v.z;
        dst[i + 3] = (bf16_t)v.w;
    }
}

// W [K x N=4096] fp32 row-major -> Wt [N x K] bf16 row-major (LDS-tiled)
__global__ __launch_bounds__(256)
void transpose_cvt_kernel(const float* __restrict__ W,
                          bf16_t* __restrict__ Wt, int K, int N) {
    __shared__ bf16_t tile[32][33];
    const int nb = blockIdx.x * 32;
    const int kb = blockIdx.y * 32;
    const int tx = threadIdx.x & 31;
    const int ty = threadIdx.x >> 5; // 0..7
#pragma unroll
    for (int i = ty; i < 32; i += 8)
        tile[i][tx] = (bf16_t)W[(size_t)(kb + i) * N + (nb + tx)];
    __syncthreads();
#pragma unroll
    for (int i = ty; i < 32; i += 8)
        Wt[(size_t)(nb + i) * K + (kb + tx)] = tile[tx][i];
}

// zero h buffers (buf 0) and c states
__global__ __launch_bounds__(256)
void init_state_kernel(bf16_t* h0bf, bf16_t* h1bf, float* c0, float* c1) {
    int i = blockIdx.x * 256 + threadIdx.x;
    if (i < BATCH_ * NHID_) {
        h0bf[i] = (bf16_t)0.0f;
        h1bf[i] = (bf16_t)0.0f;
        c0[i] = 0.0f;
        c1[i] = 0.0f;
    }
}

__global__ __launch_bounds__(256)
void copy_f32_kernel(const float* __restrict__ src, float* __restrict__ dst, int n) {
    int i = blockIdx.x * 256 + threadIdx.x;
    if (i < n) dst[i] = src[i];
}

// ---------------------------------------------------------------------------
extern "C" void kernel_launch(void* const* d_in, const int* in_sizes, int n_in,
                              void* d_out, int out_size, void* d_ws, size_t ws_size,
                              hipStream_t stream) {
    const float* x  = (const float*)d_in[0];  // [SEQ, B, NIN]
    const float* W0 = (const float*)d_in[1];  // [NIN+NHID, 4*NHID]
    const float* b0 = (const float*)d_in[2];  // [4*NHID]
    const float* W1 = (const float*)d_in[3];  // [2*NHID, 4*NHID]
    const float* b1 = (const float*)d_in[4];  // [4*NHID]
    float* out = (float*)d_out;               // [SEQ*B*H] ++ [2*B*H]

    const int K0a = NIN_ + NHID_;   // 1536
    const int K1a = 2 * NHID_;      // 2048
    const size_t BH = (size_t)BATCH_ * NHID_;

    char* ws = (char*)d_ws;
    bf16_t* x_bf = (bf16_t*)ws; ws += (size_t)SEQ_ * BATCH_ * NIN_ * sizeof(bf16_t);
    bf16_t* W0t  = (bf16_t*)ws; ws += (size_t)NGATE_ * K0a * sizeof(bf16_t);
    bf16_t* W1t  = (bf16_t*)ws; ws += (size_t)NGATE_ * K1a * sizeof(bf16_t);
    bf16_t* h0bf = (bf16_t*)ws; ws += 2 * BH * sizeof(bf16_t);  // double buffer
    bf16_t* h1bf = (bf16_t*)ws; ws += 2 * BH * sizeof(bf16_t);  // double buffer
    float*  c0   = (float*)ws;  ws += BH * sizeof(float);
    float*  c1   = (float*)ws;  ws += BH * sizeof(float);
    float*  hscr = (float*)ws;  ws += BH * sizeof(float);       // layer0 f32 sink

    // --- prep: convert x, transpose+convert weights, zero state ---
    {
        int n = SEQ_ * BATCH_ * NIN_;
        cvt_f32_bf16_kernel<<<n / (256 * 4), 256, 0, stream>>>(x, x_bf, n);
    }
    transpose_cvt_kernel<<<dim3(NGATE_ / 32, K0a / 32), 256, 0, stream>>>(W0, W0t, K0a, NGATE_);
    transpose_cvt_kernel<<<dim3(NGATE_ / 32, K1a / 32), 256, 0, stream>>>(W1, W1t, K1a, NGATE_);
    init_state_kernel<<<(int)(BH / 256), 256, 0, stream>>>(h0bf, h1bf, c0, c1);

    // --- recurrence ---
    const int ngrid = NHID_ / 16; // 64 workgroups
    for (int t = 0; t < SEQ_; ++t) {
        const int rb = t & 1;        // read buffer
        const int wb = (t + 1) & 1;  // write buffer
        // layer 0: A = concat(x_t [B,512], h0_prev [B,1024])
        float* h0f32 = (t == SEQ_ - 1) ? (out + (size_t)SEQ_ * BH) : hscr;
        lstm_cell_kernel<<<ngrid, 256, 0, stream>>>(
            x_bf + (size_t)t * BATCH_ * NIN_, NIN_,
            h0bf + (size_t)rb * BH, NHID_,
            W0t, b0, c0,
            h0bf + (size_t)wb * BH, h0f32);
        // layer 1: A = concat(h0_new [B,1024], h1_prev [B,1024])
        lstm_cell_kernel<<<ngrid, 256, 0, stream>>>(
            h0bf + (size_t)wb * BH, NHID_,
            h1bf + (size_t)rb * BH, NHID_,
            W1t, b1, c1,
            h1bf + (size_t)wb * BH, out + (size_t)t * BH);
    }

    // h_final[1] = layer1 h at t = SEQ-1 (same as last output slice)
    copy_f32_kernel<<<(int)(BH / 256), 256, 0, stream>>>(
        out + (size_t)(SEQ_ - 1) * BH, out + (size_t)SEQ_ * BH + BH, (int)BH);
}
